// CnnWordSeg_63823214018765
// MI455X (gfx1250) — matrix operand order repaired and checked by reading, not compiled, not run on hardware
//
#include <hip/hip_runtime.h>
#include <hip/hip_bf16.h>

typedef __attribute__((ext_vector_type(16))) _Float16 v16h;
typedef __attribute__((ext_vector_type(8)))  _Float16 v8h;
typedef __attribute__((ext_vector_type(4)))  _Float16 v4h;
typedef __attribute__((ext_vector_type(8)))  float    v8f;

#define BB 128
#define TT 512
#define CC 256

// LDS activation tile: 130 rows (t-1 .. t+128, clamped) x 256 ci f16,
// row stride padded to 528 B (264 halves) -> 4-bank row step, 16B aligned.
#define ROWS    130
#define RSTRIDE 264

// ---------------------------------------------------------------------------
// Embedding gather: x[B,T] -> h[B][T][C] f16 (time-major). 4 ch / thread.
// ---------------------------------------------------------------------------
__global__ void k_embed(const int* __restrict__ x, const float* __restrict__ emb,
                        _Float16* __restrict__ out) {
    int i  = blockIdx.x * blockDim.x + threadIdx.x;   // [0, B*T*64)
    int c4 = (i & 63) << 2;
    int bt = i >> 6;
    int idx = x[bt];
    const float* src = emb + (size_t)idx * CC + c4;
    float4 v = *(const float4*)src;
    v4h o;
    o[0] = (_Float16)v.x; o[1] = (_Float16)v.y;
    o[2] = (_Float16)v.z; o[3] = (_Float16)v.w;
    *(v4h*)(out + (size_t)bt * CC + c4) = o;
}

// ---------------------------------------------------------------------------
// Pack conv weights [O=256][I=256][K=3] fp32 -> WMMA A-fragment order (f16):
// apack[tap][co_blk(16)][ci_blk(8)][lane(32)][elem(16)]
// 16x32 f16 A layout: lane<16: M=lane,  e<8 -> K=e,   e>=8 -> K=e+8
//                     lane>=16: M=lane-16, e<8 -> K=e+8, e>=8 -> K=e+16
// ---------------------------------------------------------------------------
__global__ void k_pack(const float* __restrict__ w, _Float16* __restrict__ apack) {
    int p    = blockIdx.x * blockDim.x + threadIdx.x;  // [0, 196608)
    int e    = p & 15;
    int lane = (p >> 4) & 31;
    int cib  = (p >> 9) & 7;
    int cob  = (p >> 12) & 15;
    int tap  = p >> 16;
    int m = lane & 15;
    int k;
    if (lane < 16) k = (e < 8) ? e : (e + 8);
    else           k = (e < 8) ? (e + 8) : (e + 16);
    int co = cob * 16 + m;
    int ci = cib * 32 + k;
    apack[p] = (_Float16)w[((size_t)co * CC + ci) * 3 + tap];
}

// ---------------------------------------------------------------------------
// Conv1d(k=3, replicate pad) + bias + ReLU as WMMA GEMM with LDS staging.
// Block (256 thr = 8 waves) computes [1 batch] x [128 t] x [256 co].
//  - 130x256 f16 input window async-copied to LDS (ASYNCcnt + barrier).
//  - wave wv covers co-blocks {2wv, 2wv+1}, all 8 t-subtiles: 8x2 register
//    blocking, 16 f32 accumulators, 384 WMMAs/wave; each A fragment (weights,
//    L2) reused 8x -> 128 B/wave of L2 traffic per WMMA.
// Grid: 128 b * 4 t-groups = 512 blocks.
// ---------------------------------------------------------------------------
__global__ void __launch_bounds__(256)
k_conv(const _Float16* __restrict__ in, const _Float16* __restrict__ apack,
       const float* __restrict__ bias, _Float16* __restrict__ out) {
    __shared__ __align__(32) _Float16 smem[ROWS * RSTRIDE];

    const int tid = threadIdx.x;
    const int b   = blockIdx.x >> 2;
    const int tg0 = (blockIdx.x & 3) * 128;

    // ---- stage activation window into LDS via async copies ----------------
    // 130 rows * 512 B = 4160 chunks of 16 B; 256 threads -> up to 17 each.
    {
        const char* gbase = (const char*)(in + (size_t)b * TT * CC);
        uint32_t lds0 = (uint32_t)(uintptr_t)(void*)smem;
#pragma unroll
        for (int k = 0; k < 17; ++k) {
            int c = tid + k * 256;
            if (c < ROWS * 32) {
                int row = c >> 5;
                int col = (c & 31) * 16;                 // byte column
                int ts  = tg0 - 1 + row;
                ts = ts < 0 ? 0 : (ts > TT - 1 ? TT - 1 : ts);  // replicate pad
                uint64_t ga = (uint64_t)(uintptr_t)(gbase + (size_t)ts * (CC * 2) + col);
                uint32_t la = lds0 + row * (RSTRIDE * 2) + col;
                asm volatile("global_load_async_to_lds_b128 %0, %1, off"
                             :: "v"(la), "v"(ga) : "memory");
            }
        }
        asm volatile("s_wait_asynccnt 0x0" ::: "memory");
    }
    __syncthreads();

    // ---- WMMA main loop ---------------------------------------------------
    const int lane = tid & 31;
    const int wv   = tid >> 5;            // 0..7 -> co-blocks 2wv, 2wv+1
    const int tN   = lane & 15;           // column (time) within 16-wide tile
    const int hi   = lane >> 4;           // B rows 0..7,16..23 / 8..15,24..31 ; D rows 0..7 / 8..15
    const int ci_half = hi * 8;           // element offset into ci chunk

    v8f acc[8][2] = {};

#pragma unroll
    for (int tap = 0; tap < 3; ++tap) {
#pragma unroll
        for (int cib = 0; cib < 8; ++cib) {
            const _Float16* a0p = apack +
                ((((size_t)tap * 16 + 2 * wv) * 8 + cib) * 32 + lane) * 16;
            v16h a0 = *(const v16h*)a0p;
            v16h a1 = *(const v16h*)(a0p + 8 * 32 * 16);   // next co-block

#pragma unroll
            for (int tsub = 0; tsub < 8; ++tsub) {
                int r = tsub * 16 + tN + tap;              // LDS row (pre-clamped)
                const _Float16* bp = smem + (size_t)r * RSTRIDE + cib * 32 + ci_half;
                v8h lo  = *(const v8h*)bp;
                v8h hi8 = *(const v8h*)(bp + 16);
                v16h bf = __builtin_shufflevector(lo, hi8,
                            0,1,2,3,4,5,6,7,8,9,10,11,12,13,14,15);
                acc[tsub][0] = __builtin_amdgcn_wmma_f32_16x16x32_f16(
                    false, a0, false, bf, (short)0, acc[tsub][0], false, false);
                acc[tsub][1] = __builtin_amdgcn_wmma_f32_16x16x32_f16(
                    false, a1, false, bf, (short)0, acc[tsub][1], false, false);
            }
        }
    }

    // ---- epilogue: bias + ReLU + f16 pack, one b128 store per tile --------
#pragma unroll
    for (int cp = 0; cp < 2; ++cp) {
        int co_base = (2 * wv + cp) * 16 + hi * 8;
        float4 bsa = *(const float4*)(bias + co_base);
        float4 bsb = *(const float4*)(bias + co_base + 4);
        float bs[8] = {bsa.x, bsa.y, bsa.z, bsa.w, bsb.x, bsb.y, bsb.z, bsb.w};
#pragma unroll
        for (int tsub = 0; tsub < 8; ++tsub) {
            int t = tg0 + tsub * 16 + tN;
            v8h res;
#pragma unroll
            for (int r = 0; r < 8; ++r) {
                float v = acc[tsub][cp][r] + bs[r];
                res[r] = (_Float16)(v > 0.f ? v : 0.f);
            }
            *(v8h*)(out + ((size_t)b * TT + t) * CC + co_base) = res;
        }
    }
}

// ---------------------------------------------------------------------------
// Dense projection to 4 tags: em[b,t,l] = h[b,t,:] . dw[l,:] + db[l]
// ---------------------------------------------------------------------------
__global__ void k_dense(const _Float16* __restrict__ h, const float* __restrict__ dw,
                        const float* __restrict__ db, float* __restrict__ em) {
    int bt = blockIdx.x * blockDim.x + threadIdx.x;
    const _Float16* src = h + (size_t)bt * CC;
    float a0 = db[0], a1 = db[1], a2 = db[2], a3 = db[3];
#pragma unroll 8
    for (int c = 0; c < CC; ++c) {
        float v = (float)src[c];
        a0 += v * dw[c];
        a1 += v * dw[CC + c];
        a2 += v * dw[2 * CC + c];
        a3 += v * dw[3 * CC + c];
    }
    float4 o = make_float4(a0, a1, a2, a3);
    *(float4*)(em + (size_t)bt * 4) = o;
}

// ---------------------------------------------------------------------------
// CRF sum log-likelihood. One lane per batch row; sequential over T.
// ---------------------------------------------------------------------------
__global__ void k_crf(const float* __restrict__ em, const int* __restrict__ y,
                      const unsigned char* __restrict__ mask,
                      const float* __restrict__ st, const float* __restrict__ etr,
                      const float* __restrict__ tr, float* __restrict__ outp) {
    __shared__ float red[128];
    int b = threadIdx.x;

    float tm[16];
#pragma unroll
    for (int i = 0; i < 16; ++i) tm[i] = tr[i];

    const float* e  = em + (size_t)b * TT * 4;
    const int* yb   = y + (size_t)b * TT;
    const unsigned char* mb = mask + (size_t)b * TT;

    int   prev = yb[0];
    float num  = st[prev] + e[prev];
    float alpha[4];
#pragma unroll
    for (int j = 0; j < 4; ++j) alpha[j] = st[j] + e[j];

    for (int t = 1; t < TT; ++t) {
        const float* ev = e + t * 4;
        bool m = mb[t] != 0;
        int tg = yb[t];
        if (m) { num += tr[prev * 4 + tg] + ev[tg]; prev = tg; }

        float nxt[4];
#pragma unroll
        for (int j = 0; j < 4; ++j) {
            float x0 = alpha[0] + tm[0 + j];
            float x1 = alpha[1] + tm[4 + j];
            float x2 = alpha[2] + tm[8 + j];
            float x3 = alpha[3] + tm[12 + j];
            float mx = fmaxf(fmaxf(x0, x1), fmaxf(x2, x3));
            nxt[j] = mx + logf(expf(x0 - mx) + expf(x1 - mx) +
                               expf(x2 - mx) + expf(x3 - mx)) + ev[j];
        }
        if (m) {
#pragma unroll
            for (int j = 0; j < 4; ++j) alpha[j] = nxt[j];
        }
    }
    num += etr[prev];

    float z0 = alpha[0] + etr[0], z1 = alpha[1] + etr[1];
    float z2 = alpha[2] + etr[2], z3 = alpha[3] + etr[3];
    float mx = fmaxf(fmaxf(z0, z1), fmaxf(z2, z3));
    float logz = mx + logf(expf(z0 - mx) + expf(z1 - mx) + expf(z2 - mx) + expf(z3 - mx));

    red[b] = num - logz;
    __syncthreads();
    if (b == 0) {
        float s = 0.f;
        for (int i = 0; i < 128; ++i) s += red[i];
        outp[0] = s;
    }
}

// ---------------------------------------------------------------------------
extern "C" void kernel_launch(void* const* d_in, const int* in_sizes, int n_in,
                              void* d_out, int out_size, void* d_ws, size_t ws_size,
                              hipStream_t stream) {
    const int*   x    = (const int*)d_in[0];
    const int*   y    = (const int*)d_in[1];
    const unsigned char* mask = (const unsigned char*)d_in[2];   // jnp.bool_ = 1 byte
    const float* emb  = (const float*)d_in[3];
    const float* w1   = (const float*)d_in[4];
    const float* b1   = (const float*)d_in[5];
    const float* w2   = (const float*)d_in[6];
    const float* b2   = (const float*)d_in[7];
    const float* w3   = (const float*)d_in[8];
    const float* b3   = (const float*)d_in[9];
    const float* dw   = (const float*)d_in[10];
    const float* db   = (const float*)d_in[11];
    const float* st   = (const float*)d_in[12];
    const float* etr  = (const float*)d_in[13];
    const float* tr   = (const float*)d_in[14];

    char* ws = (char*)d_ws;
    const size_t ACT   = (size_t)BB * TT * CC * sizeof(_Float16);  // 33,554,432
    const size_t APACK = (size_t)3 * CC * CC * sizeof(_Float16);   // 393,216
    _Float16* hA  = (_Float16*)ws;
    _Float16* hB  = (_Float16*)(ws + ACT);
    _Float16* ap1 = (_Float16*)(ws + 2 * ACT);
    _Float16* ap2 = (_Float16*)(ws + 2 * ACT + APACK);
    _Float16* ap3 = (_Float16*)(ws + 2 * ACT + 2 * APACK);
    float*    em  = (float*)   (ws + 2 * ACT + 3 * APACK);

    k_embed<<<16384, 256, 0, stream>>>(x, emb, hA);
    k_pack <<<  768, 256, 0, stream>>>(w1, ap1);
    k_pack <<<  768, 256, 0, stream>>>(w2, ap2);
    k_pack <<<  768, 256, 0, stream>>>(w3, ap3);
    k_conv <<<  512, 256, 0, stream>>>(hA, ap1, b1, hB);
    k_conv <<<  512, 256, 0, stream>>>(hB, ap2, b2, hA);
    k_conv <<<  512, 256, 0, stream>>>(hA, ap3, b3, hB);
    k_dense<<<  256, 256, 0, stream>>>(hB, dw, db, em);
    k_crf  <<<    1, 128, 0, stream>>>(em, y, mask, st, etr, tr, (float*)d_out);
}
